// MultiHeadAttentionLayer_14027363189318
// MI455X (gfx1250) — compile-verified, matrix operand order checked by reading, not run on hardware
//
#include <hip/hip_runtime.h>

typedef _Float16 h16;
typedef __attribute__((ext_vector_type(16))) _Float16 v16h;
typedef __attribute__((ext_vector_type(8)))  _Float16 v8h;
typedef __attribute__((ext_vector_type(4)))  _Float16 v4h;
typedef __attribute__((ext_vector_type(8)))  float    v8f;
typedef __attribute__((ext_vector_type(4)))  float    v4f;

#define NNODES 3072
#define HIDDIM 256
#define NHEADS 8
#define DHEAD  32
#define DEG    16
#define EDGEDIM 32
#define FFNDIM 1024

// ---------------------------------------------------------------------------
// WMMA GEMM body:  C[m0:m0+64, n0:n0+64] = act( A @ W + bias ) (+ resid)
// 256 threads = 8 waves in a 4(M) x 2(N) grid; each wave owns 16x32 of C
// (2 x v8f accumulators). k-step 64, f16 compute / f32 accumulate via
// v_wmma_f32_16x16x32_f16. Register-pipelined global->LDS staging.
// ---------------------------------------------------------------------------
template <bool RELU, bool HAS_RESID>
__device__ __forceinline__ void gemm_body(
    const float* __restrict__ A, const float* __restrict__ W,
    const float* __restrict__ bias, const float* __restrict__ resid,
    float* __restrict__ C, int N, int K, int m0, int n0) {
  // Row stride 72 halves (144B = 36 banks) -> conflict-free fragment reads.
  __shared__ __align__(16) h16 As[64][72];
  __shared__ __align__(16) h16 Bs[64][72];

  const int tid  = threadIdx.x;
  const int wave = tid >> 5;
  const int lane = tid & 31;
  const int rw   = (wave >> 1) * 16;  // wave's row offset in tile
  const int cw   = (wave & 1) * 32;   // wave's col offset in tile
  const int r    = lane & 15;
  const int hig  = lane >> 4;         // half-wave group
  const int hi   = hig * 8;           // fragment k-offset

  v8f acc[2] = {};
  v4f ra[4], rb[4];

  // ---- preload first k-slab (64 deep) into registers ----
#pragma unroll
  for (int i = 0; i < 4; ++i) {
    int idx = tid + i * 256;
    int ar = idx >> 4, ac = (idx & 15) * 4;
    ra[i] = *(const v4f*)(A + (size_t)(m0 + ar) * K + ac);
    rb[i] = *(const v4f*)(W + (size_t)ar * N + n0 + ac);
  }

  for (int kk = 0; kk < K; kk += 64) {
    // ---- commit registers to LDS (f32 -> f16); B stored transposed ----
#pragma unroll
    for (int i = 0; i < 4; ++i) {
      int idx = tid + i * 256;
      int ar = idx >> 4, ac = (idx & 15) * 4;
      v4h ah = {(h16)ra[i].x, (h16)ra[i].y, (h16)ra[i].z, (h16)ra[i].w};
      *(v4h*)(&As[ar][ac]) = ah;
      Bs[ac + 0][ar] = (h16)rb[i].x;
      Bs[ac + 1][ar] = (h16)rb[i].y;
      Bs[ac + 2][ar] = (h16)rb[i].z;
      Bs[ac + 3][ar] = (h16)rb[i].w;
    }
    __syncthreads();

    // ---- issue next slab's global loads (overlap with WMMA below) ----
    if (kk + 64 < K) {
#pragma unroll
      for (int i = 0; i < 4; ++i) {
        int idx = tid + i * 256;
        int ar = idx >> 4, ac = (idx & 15) * 4;
        ra[i] = *(const v4f*)(A + (size_t)(m0 + ar) * K + kk + 64 + ac);
        rb[i] = *(const v4f*)(W + (size_t)(kk + 64 + ar) * N + n0 + ac);
      }
      if (kk + 128 < K) {  // pull the slab after that toward the caches
        __builtin_prefetch(A + (size_t)(m0 + (tid >> 2)) * K + kk + 128, 0, 3);
        __builtin_prefetch(W + (size_t)(kk + 128 + (tid >> 2)) * N + n0, 0, 3);
      }
    }

    // ---- 4 WMMAs per wave from LDS ----
#pragma unroll
    for (int kh = 0; kh < 2; ++kh) {
      const int ko = kh * 32;
      v8h alo = *(const v8h*)(&As[rw + r][ko + hi]);
      v8h ahi = *(const v8h*)(&As[rw + r][ko + 16 + hi]);
      v16h af = __builtin_shufflevector(alo, ahi, 0, 1, 2, 3, 4, 5, 6, 7, 8, 9,
                                        10, 11, 12, 13, 14, 15);
#pragma unroll
      for (int nt = 0; nt < 2; ++nt) {
        v8h blo = *(const v8h*)(&Bs[cw + nt * 16 + r][ko + hi]);
        v8h bhi = *(const v8h*)(&Bs[cw + nt * 16 + r][ko + 16 + hi]);
        v16h bf = __builtin_shufflevector(blo, bhi, 0, 1, 2, 3, 4, 5, 6, 7, 8,
                                          9, 10, 11, 12, 13, 14, 15);
        acc[nt] = __builtin_amdgcn_wmma_f32_16x16x32_f16(
            false, af, false, bf, (short)0, acc[nt], false, false);
      }
    }
    __syncthreads();
  }

  // ---- epilogue: C/D layout: col = lane&15, row = v + 8*(lane>>4) ----
#pragma unroll
  for (int nt = 0; nt < 2; ++nt) {
    int col  = n0 + cw + nt * 16 + r;
    float bv = bias[col];
#pragma unroll
    for (int v = 0; v < 8; ++v) {
      int row = m0 + rw + v + 8 * hig;
      float x = acc[nt][v] + bv;
      if (RELU) x = fmaxf(x, 0.0f);
      if (HAS_RESID) x += resid[(size_t)row * N + col];
      C[(size_t)row * N + col] = x;
    }
  }
}

template <bool RELU, bool HAS_RESID>
__global__ __launch_bounds__(256) void gemm_wmma_kernel(
    const float* __restrict__ A, const float* __restrict__ W,
    const float* __restrict__ bias, const float* __restrict__ resid,
    float* __restrict__ C, int N, int K) {
  gemm_body<RELU, HAS_RESID>(A, W, bias, resid, C, N, K, blockIdx.x * 64,
                             blockIdx.y * 64);
}

// Fused QKV: blockIdx.z selects {Wq,bq,Q} / {Wk,bk,K} / {Wv,bv,V}.
__global__ __launch_bounds__(256) void gemm_qkv_kernel(
    const float* __restrict__ X, const float* __restrict__ Wq,
    const float* __restrict__ Wk, const float* __restrict__ Wv,
    const float* __restrict__ bq, const float* __restrict__ bk,
    const float* __restrict__ bv, float* __restrict__ Qb,
    float* __restrict__ Kb, float* __restrict__ Vb) {
  const int z = blockIdx.z;
  const float* W = (z == 0) ? Wq : (z == 1) ? Wk : Wv;
  const float* b = (z == 0) ? bq : (z == 1) ? bk : bv;
  float* C       = (z == 0) ? Qb : (z == 1) ? Kb : Vb;
  gemm_body<false, false>(X, W, b, nullptr, C, HIDDIM, HIDDIM,
                          blockIdx.x * 64, blockIdx.y * 64);
}

// ---------------------------------------------------------------------------
// Sparse graph attention: one node per wave32.
// Lanes: j = lane&15 (neighbor slot), hg = lane>>4 (head group of 4 heads).
// Dedup matches the dense [H,N,N] scatter semantics (last slot wins).
// ---------------------------------------------------------------------------
__global__ __launch_bounds__(256) void attn_kernel(
    const float* __restrict__ Q, const float* __restrict__ Kp,
    const float* __restrict__ V, const int* __restrict__ nbr,
    const float* __restrict__ EF, const float* __restrict__ We,
    const float* __restrict__ be, float* __restrict__ Out) {
  __shared__ float wLDS[8][DEG][NHEADS];
  const int tid  = threadIdx.x;
  const int wave = tid >> 5;
  const int lane = tid & 31;
  const int n    = blockIdx.x * 8 + wave;
  const int j    = lane & 15;
  const int hg   = lane >> 4;
  const int* nb  = nbr + (size_t)n * DEG;
  const int mj   = nb[j];

  // a slot is shadowed if a later slot holds the same source index
  bool valid = true;
#pragma unroll
  for (int k2 = 0; k2 < DEG; ++k2)
    if (k2 > j && nb[k2] == mj) valid = false;

  const float NEG_INF = -__builtin_inff();
  float s[4];
#pragma unroll
  for (int h4 = 0; h4 < 4; ++h4) {
    int h = hg * 4 + h4;
    const float* qp = Q + (size_t)n * HIDDIM + h * DHEAD;
    const float* kp = Kp + (size_t)mj * HIDDIM + h * DHEAD;
    float dot = 0.0f;
#pragma unroll
    for (int d = 0; d < DHEAD; d += 4) {
      v4f q4 = *(const v4f*)(qp + d);
      v4f k4 = *(const v4f*)(kp + d);
      dot += q4.x * k4.x + q4.y * k4.y + q4.z * k4.z + q4.w * k4.w;
    }
    const float* ef = EF + (size_t)(n * DEG + j) * EDGEDIM;
    float ebias = be[h];
#pragma unroll
    for (int c = 0; c < EDGEDIM; ++c) ebias += ef[c] * We[c * NHEADS + h];
    s[h4] = dot * 0.17677669529663687f + ebias;  // 1/sqrt(Dh)
    if (!valid) s[h4] = NEG_INF;
  }

  // softmax over the 16 neighbor slots (within each half-wave)
#pragma unroll
  for (int h4 = 0; h4 < 4; ++h4) {
    float mx = s[h4];
    for (int m2 = 1; m2 < 16; m2 <<= 1) mx = fmaxf(mx, __shfl_xor(mx, m2, 16));
    float e = valid ? __expf(s[h4] - mx) : 0.0f;
    float sm = e;
    for (int m2 = 1; m2 < 16; m2 <<= 1) sm += __shfl_xor(sm, m2, 16);
    wLDS[wave][j][hg * 4 + h4] = e / sm;
  }
  __syncthreads();

  // phase 2: lane owns 8 output dims (all within one head)
  const int d0 = lane * 8;
  const int h  = d0 >> 5;
  v4f acc0 = {0.f, 0.f, 0.f, 0.f}, acc1 = {0.f, 0.f, 0.f, 0.f};
  for (int k2 = 0; k2 < DEG; ++k2) {
    int m    = nb[k2];
    float wk = wLDS[wave][k2][h];
    const float* vp = V + (size_t)m * HIDDIM + d0;
    v4f v0 = *(const v4f*)(vp);
    v4f v1 = *(const v4f*)(vp + 4);
    acc0 += wk * v0;
    acc1 += wk * v1;
  }
  float* op = Out + (size_t)n * HIDDIM + d0;
  *(v4f*)(op)     = acc0;
  *(v4f*)(op + 4) = acc1;
}

// ---------------------------------------------------------------------------
// Row LayerNorm over 256 cols: one row per wave32 (8 floats/lane).
// ---------------------------------------------------------------------------
__global__ __launch_bounds__(256) void ln_kernel(const float* __restrict__ Y,
                                                 const float* __restrict__ g,
                                                 const float* __restrict__ b,
                                                 float* __restrict__ O) {
  const int tid  = threadIdx.x;
  const int wave = tid >> 5;
  const int lane = tid & 31;
  const int row  = blockIdx.x * 8 + wave;
  const float* y = Y + (size_t)row * HIDDIM + lane * 8;
  float x[8];
  v4f x0 = *(const v4f*)y, x1 = *(const v4f*)(y + 4);
  x[0] = x0.x; x[1] = x0.y; x[2] = x0.z; x[3] = x0.w;
  x[4] = x1.x; x[5] = x1.y; x[6] = x1.z; x[7] = x1.w;
  float sum = 0.f;
#pragma unroll
  for (int i = 0; i < 8; ++i) sum += x[i];
  for (int m = 1; m < 32; m <<= 1) sum += __shfl_xor(sum, m, 32);
  float mu = sum * (1.0f / HIDDIM);
  float vs = 0.f;
#pragma unroll
  for (int i = 0; i < 8; ++i) { float t = x[i] - mu; vs += t * t; }
  for (int m = 1; m < 32; m <<= 1) vs += __shfl_xor(vs, m, 32);
  float rs = rsqrtf(vs * (1.0f / HIDDIM) + 1e-5f);
  const int c0 = lane * 8;
#pragma unroll
  for (int i = 0; i < 8; ++i)
    O[(size_t)row * HIDDIM + c0 + i] = (x[i] - mu) * rs * g[c0 + i] + b[c0 + i];
}

// ---------------------------------------------------------------------------
extern "C" void kernel_launch(void* const* d_in, const int* in_sizes, int n_in,
                              void* d_out, int out_size, void* d_ws,
                              size_t ws_size, hipStream_t stream) {
  const float* X    = (const float*)d_in[0];   // node_states [N,256]
  const int*   nbr  = (const int*)d_in[1];     // neighbors [N,16]
  const float* EF   = (const float*)d_in[2];   // edge_features [E,32]
  const float* Wq   = (const float*)d_in[3];
  const float* bq   = (const float*)d_in[4];
  const float* Wk   = (const float*)d_in[5];
  const float* bk   = (const float*)d_in[6];
  const float* Wv   = (const float*)d_in[7];
  const float* bv   = (const float*)d_in[8];
  const float* We   = (const float*)d_in[9];
  const float* be   = (const float*)d_in[10];
  const float* Wo   = (const float*)d_in[11];
  const float* bo   = (const float*)d_in[12];
  const float* ln1g = (const float*)d_in[13];
  const float* ln1b = (const float*)d_in[14];
  const float* ln2g = (const float*)d_in[15];
  const float* ln2b = (const float*)d_in[16];
  const float* Wf1  = (const float*)d_in[17];
  const float* bf1  = (const float*)d_in[18];
  const float* Wf2  = (const float*)d_in[19];
  const float* bf2  = (const float*)d_in[20];

  const size_t NH = (size_t)NNODES * HIDDIM;  // 786432 floats
  float* ws  = (float*)d_ws;
  float* Qb  = ws;
  float* Kb  = ws + NH;
  float* Vb  = ws + 2 * NH;
  float* AO  = ws + 3 * NH;                   // attention output
  float* T1  = ws + 4 * NH;                   // pre-LN1
  float* H1  = ws + 5 * NH;                   // post-LN1
  float* FF  = ws + 6 * NH;                   // FFN hidden [N,1024]
  float* T2  = FF + (size_t)NNODES * FFNDIM;  // pre-LN2

  dim3 blk(256);
  dim3 g_qkv(NNODES / 64, HIDDIM / 64, 3);    // 48 x 4 x 3
  dim3 g_small(NNODES / 64, HIDDIM / 64);     // 48 x 4
  dim3 g_ffn1(NNODES / 64, FFNDIM / 64);      // 48 x 16
  dim3 g_rows(NNODES / 8);                    // 384

  // fused QKV projections
  gemm_qkv_kernel<<<g_qkv, blk, 0, stream>>>(X, Wq, Wk, Wv, bq, bk, bv, Qb, Kb,
                                             Vb);
  // sparse attention over 16 neighbors, fused edge bias + softmax + gather
  attn_kernel<<<g_rows, blk, 0, stream>>>(Qb, Kb, Vb, nbr, EF, We, be, AO);
  // output projection + residual, then LN1
  gemm_wmma_kernel<false, true><<<g_small, blk, 0, stream>>>(AO, Wo, bo, X, T1,
                                                             HIDDIM, HIDDIM);
  ln_kernel<<<g_rows, blk, 0, stream>>>(T1, ln1g, ln1b, H1);
  // FFN: 256 -> 1024 (ReLU) -> 256 (+residual), then LN2 into d_out
  gemm_wmma_kernel<true, false><<<g_ffn1, blk, 0, stream>>>(H1, Wf1, bf1,
                                                            nullptr, FF,
                                                            FFNDIM, HIDDIM);
  gemm_wmma_kernel<false, true><<<g_small, blk, 0, stream>>>(FF, Wf2, bf2, H1,
                                                             T2, HIDDIM,
                                                             FFNDIM);
  ln_kernel<<<g_rows, blk, 0, stream>>>(T2, ln2g, ln2b, (float*)d_out);
}
